// SelfAttention_63453846831167
// MI455X (gfx1250) — matrix operand order repaired and checked by reading, not compiled, not run on hardware
//
#include <hip/hip_runtime.h>
#include <hip/hip_bf16.h>

// ---------------------------------------------------------------------------
// Self-attention for MI455X (gfx1250): all GEMMs via v_wmma_f32_16x16x32_bf16
// B=4, N=2048, D=1024, H=8, Hd=128
// ---------------------------------------------------------------------------

#define BATCH 4
#define SEQ   2048
#define DIM   1024
#define HEADS 8
#define HDIM  128
#define D3    (3 * DIM)

typedef __attribute__((ext_vector_type(16))) __bf16 v16bf;
typedef __attribute__((ext_vector_type(8)))  __bf16 v8bf;
typedef __attribute__((ext_vector_type(4)))  __bf16 v4bf;
typedef __attribute__((ext_vector_type(8)))  float  v8f;
typedef __attribute__((ext_vector_type(4)))  float  v4f;
typedef __attribute__((ext_vector_type(4)))  int    v4i;

// --- optional CDNA5 async global->LDS copy path (guarded; falls back) ------
#if defined(__HIP_DEVICE_COMPILE__) && \
    __has_builtin(__builtin_amdgcn_global_load_async_to_lds_b128)
#define HAVE_ASYNC_LDS 1
__device__ __forceinline__ void async_copy_b128(const void* gsrc, void* ldst) {
    // parameters are generic int4* per the toolchain's builtin signature
    __builtin_amdgcn_global_load_async_to_lds_b128((v4i*)gsrc, (v4i*)ldst, 0, 0);
}
__device__ __forceinline__ void async_wait0() {
#if __has_builtin(__builtin_amdgcn_s_wait_asynccnt)
    __builtin_amdgcn_s_wait_asynccnt(0);
#else
    asm volatile("s_wait_asynccnt 0x0" ::: "memory");
#endif
}
#else
#define HAVE_ASYNC_LDS 0
#endif

// ===========================================================================
// Tiled GEMM + bias:  C[M,N] = A[M,K] * B[K,N] + bias[N]
// A: row-major (float or bf16), B: row-major float (weights), C: float or bf16
// Block: 256 threads (8 waves), tile 128x128, K-step 32.
// Wave grid 2(M) x 4(N); each wave: 4x2 grid of 16x16 WMMA tiles.
// B tile stored TRANSPOSED in LDS so fragment reads are contiguous b128s.
// ===========================================================================
template <typename AT, typename OT>
__global__ __launch_bounds__(256) void gemm_bias_wmma(
    const AT* __restrict__ A, const float* __restrict__ Bm,
    const float* __restrict__ bias, OT* __restrict__ C,
    int M, int N, int K)
{
    __shared__ __bf16 As [128][40];   // [m][k]  (32 + 8 pad)
    __shared__ __bf16 Bst[128][40];   // [n][k]  transposed (32 + 8 pad)

    const int tid  = threadIdx.x;
    const int wave = tid >> 5;
    const int lane = tid & 31;
    const int half = lane >> 4;
    const int l16  = lane & 15;
    const int wm   = wave >> 2;      // 0..1 (M direction, 64 rows each)
    const int wn   = wave & 3;       // 0..3 (N direction, 32 cols each)

    const int row0 = blockIdx.y * 128;
    const int col0 = blockIdx.x * 128;

    v8f acc[4][2] = {};

    for (int k0 = 0; k0 < K; k0 += 32) {
        // ---- stage A tile (128x32) ----
        if constexpr (sizeof(AT) == 4) {
            // float source: float4 loads, packed 4xbf16 stores
            #pragma unroll
            for (int i = 0; i < 4; ++i) {
                int g = tid + i * 256;           // 1024 groups of 4
                int r = g >> 3, c4 = (g & 7) * 4;
                v4f v = *(const v4f*)&A[(size_t)(row0 + r) * K + k0 + c4];
                v4bf p;
                #pragma unroll
                for (int q = 0; q < 4; ++q) p[q] = (__bf16)v[q];
                *(v4bf*)&As[r][c4] = p;
            }
        } else {
            // bf16 source: straight copy -> async global->LDS when available
            #pragma unroll
            for (int i = 0; i < 2; ++i) {
                int g = tid + i * 256;           // 512 groups of 8
                int r = g >> 2, c8 = (g & 3) * 8;
#if HAVE_ASYNC_LDS
                async_copy_b128(&A[(size_t)(row0 + r) * K + k0 + c8], &As[r][c8]);
#else
                *(v8bf*)&As[r][c8] =
                    *(const v8bf*)&A[(size_t)(row0 + r) * K + k0 + c8];
#endif
            }
        }
        // ---- stage B tile (32x128) transposed: float4 loads, scatter stores ----
        #pragma unroll
        for (int i = 0; i < 4; ++i) {
            int g = tid + i * 256;               // 1024 groups of 4
            int r = g >> 5, c4 = (g & 31) * 4;   // r: k-row, c4: n-col
            v4f v = *(const v4f*)&Bm[(size_t)(k0 + r) * N + col0 + c4];
            #pragma unroll
            for (int q = 0; q < 4; ++q) Bst[c4 + q][r] = (__bf16)v[q];
        }
#if HAVE_ASYNC_LDS
        if constexpr (sizeof(AT) == 2) async_wait0();
#endif
        __syncthreads();

        // ---- B fragments hoisted: one per nt, reused across all mt ----
        v16bf bfr[2];
        #pragma unroll
        for (int nt = 0; nt < 2; ++nt) {
            const int bc = wn * 32 + nt * 16 + l16;
            const int kk = half * 16;
            #pragma unroll
            for (int j = 0; j < 16; ++j)
                bfr[nt][j] = Bst[bc][kk + j];
        }
        #pragma unroll
        for (int mt = 0; mt < 4; ++mt) {
            // A fragment: lanes 0-15 -> K 0-15 ; lanes 16-31 -> K 16-31
            v16bf af;
            const int ar  = wm * 64 + mt * 16 + l16;
            const int off = half * 16;
            #pragma unroll
            for (int j = 0; j < 8; ++j) {
                af[j]     = As[ar][off + j];
                af[j + 8] = As[ar][off + 8 + j];
            }
            #pragma unroll
            for (int nt = 0; nt < 2; ++nt)
                acc[mt][nt] = __builtin_amdgcn_wmma_f32_16x16x32_bf16(
                    false, af, false, bfr[nt], (short)0, acc[mt][nt], false, false);
        }
        __syncthreads();
    }

    // store: C layout (VGPR i -> row i for lanes 0-15, row i+8 for lanes 16-31)
    #pragma unroll
    for (int mt = 0; mt < 4; ++mt) {
        #pragma unroll
        for (int nt = 0; nt < 2; ++nt) {
            const int col = col0 + wn * 32 + nt * 16 + l16;
            const float bv = bias[col];
            #pragma unroll
            for (int i = 0; i < 8; ++i) {
                const int row = row0 + wm * 64 + mt * 16 + i + half * 8;
                C[(size_t)row * N + col] = (OT)(acc[mt][nt][i] + bv);
            }
        }
    }
}

// ===========================================================================
// Flash attention per (b, h, 128 q-rows): online softmax, WMMA for QK^T & PV.
// qkv: bf16 [B,N,3D] (Q at col 0, K at col D, V at col 2D, head h at h*Hd).
// out: bf16 [B,N,D] with heads merged.
// Block: 256 threads (8 waves); wave w owns q-rows [w*16, w*16+16).
// Key blocks of 64. Q/K staged via async global->LDS copies (when available);
// V staged TRANSPOSED ([hd][key]) so P*V B-fragments are contiguous b128s.
// ===========================================================================
__global__ __launch_bounds__(256) void flash_attn_wmma(
    const __bf16* __restrict__ qkv, __bf16* __restrict__ out)
{
    extern __shared__ char smem[];
    __bf16* Qs = (__bf16*)smem;                 // [128][136]
    __bf16* Ks = Qs + 128 * 136;                // [64][136]   (key, hd)
    __bf16* Vt = Ks + 64 * 136;                 // [128][72]   (hd, key) transposed
    __bf16* Ps = Vt + 128 * 72;                 // [8][16][72]

    const int b     = blockIdx.z;
    const int h     = blockIdx.y;
    const int qbase = blockIdx.x * 128;

    const int tid  = threadIdx.x;
    const int w    = tid >> 5;
    const int lane = tid & 31;
    const int half = lane >> 4;
    const int l16  = lane & 15;

    const float scale = 0.08838834764831845f;   // 1/sqrt(128)

    // stage Q block (128 x 128): pure copy -> async path when available
    #pragma unroll
    for (int i = 0; i < 8; ++i) {
        int g = tid + i * 256;                  // 2048 groups of 8
        int r = g >> 4, c8 = (g & 15) * 8;
        const __bf16* src = &qkv[(size_t)(b * SEQ + qbase + r) * D3 + h * HDIM + c8];
#if HAVE_ASYNC_LDS
        async_copy_b128(src, &Qs[r * 136 + c8]);
#else
        *(v8bf*)&Qs[r * 136 + c8] = *(const v8bf*)src;
#endif
    }

    float mrow[8], lrow[8];
    #pragma unroll
    for (int i = 0; i < 8; ++i) { mrow[i] = -INFINITY; lrow[i] = 0.0f; }
    v8f oacc[8] = {};

    for (int kb = 0; kb < SEQ / 64; ++kb) {
        const int kbase = kb * 64;
        __syncthreads();
        // stage K (async row-major copy) and V (transposed scatter)
        #pragma unroll
        for (int i = 0; i < 4; ++i) {
            int g = tid + i * 256;              // 1024 groups of 8
            int r = g >> 4, c8 = (g & 15) * 8;
            size_t base = (size_t)(b * SEQ + kbase + r) * D3 + h * HDIM + c8;
#if HAVE_ASYNC_LDS
            async_copy_b128(&qkv[base + DIM], &Ks[r * 136 + c8]);
#else
            *(v8bf*)&Ks[r * 136 + c8] = *(const v8bf*)&qkv[base + DIM];
#endif
            v8bf vv = *(const v8bf*)&qkv[base + 2 * DIM];
            #pragma unroll
            for (int q = 0; q < 8; ++q) Vt[(c8 + q) * 72 + r] = vv[q];
        }
#if HAVE_ASYNC_LDS
        async_wait0();                          // Q (first iter) + K complete
#endif
        __syncthreads();

        // prefetch next key block into cache while we compute on this one
        if (kb + 1 < SEQ / 64) {
            size_t nbase = (size_t)(b * SEQ + kbase + 64 + (tid >> 2)) * D3
                         + h * HDIM + (tid & 3) * 32;
            __builtin_prefetch(&qkv[nbase + DIM], 0, 1);
            __builtin_prefetch(&qkv[nbase + 2 * DIM], 0, 1);
        }

        // ---- S = Q * K^T (16 q-rows x 64 keys per wave) ----
        v8f sacc[4] = {};
        #pragma unroll
        for (int kc = 0; kc < 4; ++kc) {          // hd chunks of 32
            v16bf af;
            const int qr  = w * 16 + l16;
            const int off = kc * 32 + half * 16;
            #pragma unroll
            for (int j = 0; j < 8; ++j) {
                af[j]     = Qs[qr * 136 + off + j];
                af[j + 8] = Qs[qr * 136 + off + 8 + j];
            }
            #pragma unroll
            for (int nt = 0; nt < 4; ++nt) {      // key sub-tiles of 16
                v16bf bfr;
                const int key = nt * 16 + l16;
                const int kk  = kc * 32 + half * 16;
                #pragma unroll
                for (int j = 0; j < 16; ++j)
                    bfr[j] = Ks[key * 136 + kk + j];
                sacc[nt] = __builtin_amdgcn_wmma_f32_16x16x32_bf16(
                    false, af, false, bfr, (short)0, sacc[nt], false, false);
            }
        }

        // ---- online softmax (row reductions across 16 lanes of each half) ----
        float mnew[8], corr[8], rsum[8];
        #pragma unroll
        for (int i = 0; i < 8; ++i) {
            float mx = -INFINITY;
            #pragma unroll
            for (int nt = 0; nt < 4; ++nt) mx = fmaxf(mx, sacc[nt][i]);
            mx *= scale;
            #pragma unroll
            for (int d = 1; d < 16; d <<= 1)
                mx = fmaxf(mx, __shfl_xor(mx, d, 32));
            mnew[i] = fmaxf(mrow[i], mx);
            corr[i] = __expf(mrow[i] - mnew[i]);
            mrow[i] = mnew[i];
            rsum[i] = 0.0f;
        }
        // P = exp(S*scale - m) -> LDS (per-wave region, A-fragment source layout)
        #pragma unroll
        for (int nt = 0; nt < 4; ++nt) {
            #pragma unroll
            for (int i = 0; i < 8; ++i) {
                float p = __expf(sacc[nt][i] * scale - mnew[i]);
                rsum[i] += p;
                Ps[(w * 16 + i + half * 8) * 72 + nt * 16 + l16] = (__bf16)p;
            }
        }
        #pragma unroll
        for (int i = 0; i < 8; ++i) {
            float rs = rsum[i];
            #pragma unroll
            for (int d = 1; d < 16; d <<= 1)
                rs += __shfl_xor(rs, d, 32);
            lrow[i] = lrow[i] * corr[i] + rs;
        }
        // rescale running O
        #pragma unroll
        for (int nt = 0; nt < 8; ++nt)
            #pragma unroll
            for (int i = 0; i < 8; ++i)
                oacc[nt][i] *= corr[i];

        __syncthreads();   // P visible wave-wide before fragment reload

        // ---- O += P * V  (P: 16x64, Vt: [hd][key]) ----
        #pragma unroll
        for (int kc = 0; kc < 2; ++kc) {          // key chunks of 32
            v16bf af;
            const int pr  = w * 16 + l16;
            const int off = kc * 32 + half * 16;
            #pragma unroll
            for (int j = 0; j < 8; ++j) {
                af[j]     = Ps[pr * 72 + off + j];
                af[j + 8] = Ps[pr * 72 + off + 8 + j];
            }
            #pragma unroll
            for (int nt = 0; nt < 8; ++nt) {      // hd sub-tiles of 16
                v16bf bfr;
                const int vc = nt * 16 + l16;     // hd column
                const int kk = kc * 32 + half * 16;
                #pragma unroll
                for (int j = 0; j < 16; ++j)
                    bfr[j] = Vt[vc * 72 + kk + j];
                oacc[nt] = __builtin_amdgcn_wmma_f32_16x16x32_bf16(
                    false, af, false, bfr, (short)0, oacc[nt], false, false);
            }
        }
    }

    // ---- normalize and write out (merge heads: col = h*Hd + hd) ----
    #pragma unroll
    for (int nt = 0; nt < 8; ++nt) {
        const int col = h * HDIM + nt * 16 + l16;
        #pragma unroll
        for (int i = 0; i < 8; ++i) {
            const int row = qbase + w * 16 + i + half * 8;
            float o = oacc[nt][i] / lrow[i];
            out[(size_t)(b * SEQ + row) * DIM + col] = (__bf16)o;
        }
    }
}

// ===========================================================================
// Host launcher
// ===========================================================================
extern "C" void kernel_launch(void* const* d_in, const int* in_sizes, int n_in,
                              void* d_out, int out_size, void* d_ws, size_t ws_size,
                              hipStream_t stream) {
    const float* rep    = (const float*)d_in[0];   // [4,2048,1024]
    const float* W_qkv  = (const float*)d_in[1];   // [1024,3072]
    const float* b_qkv  = (const float*)d_in[2];   // [3072]
    const float* W_proj = (const float*)d_in[3];   // [1024,1024]
    const float* b_proj = (const float*)d_in[4];   // [1024]
    float* outp = (float*)d_out;

    const int Mrows = BATCH * SEQ;                 // 8192

    __bf16* qkv_ws  = (__bf16*)d_ws;               // 8192*3072 bf16
    __bf16* attn_ws = (__bf16*)((char*)d_ws + (size_t)Mrows * D3 * sizeof(__bf16));

    // 1) QKV projection: [8192,1024] x [1024,3072] -> bf16 qkv (+bias)
    gemm_bias_wmma<float, __bf16>
        <<<dim3(D3 / 128, Mrows / 128), 256, 0, stream>>>(
            rep, W_qkv, b_qkv, qkv_ws, Mrows, D3, DIM);

    // 2) Flash attention per (b, h, 128-row q block)
    const size_t lds_bytes =
        (128 * 136 + 64 * 136 + 128 * 72 + 8 * 16 * 72) * sizeof(__bf16); // 89088
    flash_attn_wmma
        <<<dim3(SEQ / 128, HEADS, BATCH), 256, lds_bytes, stream>>>(
            qkv_ws, attn_ws);

    // 3) Output projection: [8192,1024] x [1024,1024] -> f32 out (+bias)
    gemm_bias_wmma<__bf16, float>
        <<<dim3(DIM / 128, Mrows / 128), 256, 0, stream>>>(
            attn_ws, W_proj, b_proj, outp, Mrows, DIM, DIM);
}
